// KanLinear_31181462569454
// MI455X (gfx1250) — matrix operand order repaired and checked by reading, not compiled, not run on hardware
//
#include <hip/hip_runtime.h>

// ---------------------------------------------------------------------------
// KAN linear layer for MI455X (gfx1250, wave32, WMMA).
//   out = concat(silu(x), rbf_bases(x)) @ concat(scale_base, W2)   (K = 9216)
// Round 2: double-buffered LDS with CDNA5 async global->LDS copies
// (GLOBAL_LOAD_ASYNC_TO_LDS_B128 / ASYNCcnt) overlapping WMMA compute.
// ---------------------------------------------------------------------------

typedef __attribute__((ext_vector_type(16))) __bf16 v16bf;
typedef __attribute__((ext_vector_type(8)))  __bf16 v8bf;
typedef __attribute__((ext_vector_type(8)))  float  v8f;

#define B_SZ   8192
#define I_SZ   1024
#define G_SZ   8
#define O_SZ   1024
#define K_TOT  (I_SZ + I_SZ * G_SZ)   // 9216

#define BM 128
#define BN 128
#define BK 64

// ---------------- prep: activations -> bf16 A[B][K_TOT] ----------------
__global__ __launch_bounds__(256) void kan_prep_act(
    const float* __restrict__ x, const float* __restrict__ grid,
    const float* __restrict__ sigma, __bf16* __restrict__ A)
{
  int idx = blockIdx.x * blockDim.x + threadIdx.x;   // b*I + i
  if (idx >= B_SZ * I_SZ) return;
  int i = idx & (I_SZ - 1);
  int b = idx >> 10;
  float xv = x[idx];
  size_t rb = (size_t)b * K_TOT;
  // silu(x) = x * sigmoid(x)
  A[rb + i] = (__bf16)(xv / (1.0f + __expf(-xv)));
  float inv_sig = 1.0f / sigma[0];
  v8bf out;
#pragma unroll
  for (int g = 0; g < G_SZ; ++g) {
    float t = (xv - grid[i * G_SZ + g]) * inv_sig;
    out[g] = (__bf16)__expf(-t * t);
  }
  *(v8bf*)&A[rb + I_SZ + (size_t)i * G_SZ] = out;   // 16B aligned
}

// -------- prep: weights (scale_spline folded) -> bf16 Wt[O][K_TOT] --------
__global__ __launch_bounds__(256) void kan_prep_w(
    const float* __restrict__ scale_base, const float* __restrict__ spline_w,
    const float* __restrict__ scale_spline, __bf16* __restrict__ Wt)
{
  int idx = blockIdx.x * blockDim.x + threadIdx.x;   // o*I + i
  if (idx >= O_SZ * I_SZ) return;
  int i = idx & (I_SZ - 1);
  int o = idx >> 10;
  size_t rb = (size_t)o * K_TOT;
  Wt[rb + i] = (__bf16)scale_base[(size_t)i * O_SZ + o];
  float ss = scale_spline[(size_t)i * O_SZ + o];
  const float* w = &spline_w[(size_t)idx * G_SZ];    // [o][i][g] contiguous
  v8bf v;
#pragma unroll
  for (int g = 0; g < G_SZ; ++g) v[g] = (__bf16)(w[g] * ss);
  *(v8bf*)&Wt[rb + I_SZ + (size_t)i * G_SZ] = v;
}

// ---------------- GEMM: out[B][O] = A[B][K] * Wt[O][K]^T ----------------
__global__ __launch_bounds__(256) void kan_gemm(
    const __bf16* __restrict__ A, const __bf16* __restrict__ Wt,
    float* __restrict__ Out)
{
  // Double-buffered LDS tiles (4 x 16KB = 64KB).
  // As stores each 32-wide K group in natural order (like Bs), so a
  // lane's 16 A elements (k = 16*half + s, same as B) are contiguous.
  __shared__ __align__(32) __bf16 As[2][BM][BK];
  __shared__ __align__(32) __bf16 Bs[2][BN][BK];   // n-major: 16 contiguous K per lane

  const int tid  = threadIdx.x;
  const int lane = tid & 31;
  const int wid  = tid >> 5;            // 8 waves
  const int wm   = (wid >> 2) * 64;     // wave tile: 64 (M) x 32 (N)
  const int wn   = (wid & 3) * 32;
  const int bm   = blockIdx.y * BM;
  const int bn   = blockIdx.x * BN;
  const int lm   = lane & 15;           // row/col within 16-tile
  const int lh   = lane >> 4;           // half select

  v8f acc[4][2] = {};

  // Issue one tile's worth of async global->LDS copies (8 instrs/thread;
  // per-lane 16B, 1024 chunks per 128x64 bf16 tile across 256 threads).
  auto copy_async = [&](int k0, int buf) {
#pragma unroll
    for (int c = 0; c < 4; ++c) {
      int chunk = tid + c * 256;        // 0..1023
      int row   = chunk >> 3;           // 0..127
      int cc    = chunk & 7;            // 16B chunk within 64-half row
      int sub   = cc & 3;
      int psub  = ((sub & 1) << 1) | (sub >> 1);   // 0->0, 1->2, 2->1, 3->3
      int pcc   = (cc & 4) | psub;
      unsigned ldsA = (unsigned)(size_t)&As[buf][row][cc * 8];
      unsigned ldsB = (unsigned)(size_t)&Bs[buf][row][cc * 8];
      unsigned long long ga =
          (unsigned long long)&A[(size_t)(bm + row) * K_TOT + k0 + cc * 8];
      unsigned long long gb =
          (unsigned long long)&Wt[(size_t)(bn + row) * K_TOT + k0 + cc * 8];
      asm volatile("global_load_async_to_lds_b128 %0, %1, off"
                   :: "v"(ldsA), "v"(ga) : "memory");
      asm volatile("global_load_async_to_lds_b128 %0, %1, off"
                   :: "v"(ldsB), "v"(gb) : "memory");
    }
  };

  auto compute = [&](int buf) {
#pragma unroll
    for (int s = 0; s < 2; ++s) {       // two 32-deep WMMA K steps per stage
      v16bf bfrag[2];
#pragma unroll
      for (int nt = 0; nt < 2; ++nt)
        bfrag[nt] = *(const v16bf*)&Bs[buf][wn + nt * 16 + lm][s * 32 + lh * 16];
#pragma unroll
      for (int mt = 0; mt < 4; ++mt) {
        v16bf afrag = *(const v16bf*)&As[buf][wm + mt * 16 + lm][s * 32 + lh * 16];
#pragma unroll
        for (int nt = 0; nt < 2; ++nt)
          acc[mt][nt] = __builtin_amdgcn_wmma_f32_16x16x32_bf16(
              /*neg_a=*/false, afrag, /*neg_b=*/false, bfrag[nt],
              /*c_mod=*/(short)0, acc[mt][nt],
              /*reuse_a=*/false, /*reuse_b=*/false);
      }
    }
  };

  const int nk = K_TOT / BK;            // 144 tiles
  copy_async(0, 0);
  int buf = 0;
  for (int t = 0; t < nk; ++t) {
    // My async copies of tile t have landed in LDS[buf].
    asm volatile("s_wait_asynccnt 0x0" ::: "memory");
    // Everyone's copies landed; everyone finished reading LDS[buf^1].
    __syncthreads();
    if (t + 1 < nk) copy_async((t + 1) * BK, buf ^ 1);  // overlaps compute
    compute(buf);
    buf ^= 1;
  }

  // Epilogue: C/D layout -> lane L: n = L%16; VGPR v: m = v + 8*(L/16).
#pragma unroll
  for (int mt = 0; mt < 4; ++mt) {
#pragma unroll
    for (int nt = 0; nt < 2; ++nt) {
      int n = bn + wn + nt * 16 + lm;
#pragma unroll
      for (int v = 0; v < 8; ++v) {
        int m = bm + wm + mt * 16 + v + 8 * lh;
        Out[(size_t)m * O_SZ + n] = acc[mt][nt][v];
      }
    }
  }
}

extern "C" void kernel_launch(void* const* d_in, const int* in_sizes, int n_in,
                              void* d_out, int out_size, void* d_ws, size_t ws_size,
                              hipStream_t stream) {
  const float* x            = (const float*)d_in[0];
  const float* scale_base   = (const float*)d_in[1];
  const float* spline_w     = (const float*)d_in[2];
  const float* scale_spline = (const float*)d_in[3];
  const float* grid         = (const float*)d_in[4];
  const float* sigma        = (const float*)d_in[5];
  float* out = (float*)d_out;

  __bf16* A  = (__bf16*)d_ws;                                      // 8192*9216*2 B
  __bf16* Wt = (__bf16*)((char*)d_ws + (size_t)B_SZ * K_TOT * 2);  // 9216*1024*2 B

  kan_prep_act<<<(B_SZ * I_SZ) / 256, 256, 0, stream>>>(x, grid, sigma, A);
  kan_prep_w<<<(O_SZ * I_SZ) / 256, 256, 0, stream>>>(scale_base, spline_w,
                                                      scale_spline, Wt);
  dim3 grid_dim(O_SZ / BN, B_SZ / BM);   // 8 x 64 blocks
  kan_gemm<<<grid_dim, 256, 0, stream>>>(A, Wt, out);
}
